// SpatioTemporalAttention_85590108275000
// MI455X (gfx1250) — matrix operand order repaired and checked by reading, not compiled, not run on hardware
//
#include <hip/hip_runtime.h>
#include <hip/hip_bf16.h>

// ---------------- problem constants ----------------
#define BB       2
#define TT       16
#define SS       256          // GH*GW
#define DMODEL   1536
#define NHEADS   16
#define NGROUPS  4
#define HDIM     96
#define QKVOUT   2304         // 16*96 + 2*4*96
#define MROWS    (BB*TT*SS)   // 8192

typedef __attribute__((ext_vector_type(16))) __bf16 v16bf;
typedef __attribute__((ext_vector_type(8)))  __bf16 v8bf;
typedef __attribute__((ext_vector_type(8)))  float  v8f;
typedef __attribute__((ext_vector_type(4)))  float  v4f;

__device__ __forceinline__ __bf16 f2bf(float f) {
    unsigned u = __builtin_bit_cast(unsigned, f);
    unsigned r = u + 0x7FFFu + ((u >> 16) & 1u);       // round-to-nearest-even
    return __builtin_bit_cast(__bf16, (unsigned short)(r >> 16));
}

__device__ __forceinline__ v8f wmma_bf16(v16bf a, v16bf b, v8f c) {
    return __builtin_amdgcn_wmma_f32_16x16x32_bf16(false, a, false, b, (short)0, c,
                                                   false, false);
}

// A-fragment (16x32 bf16, contiguous K order): lane holds row M=lane&15.
// elements 0..7  <- K = 16*half + 0..7
// elements 8..15 <- K = 16*half + 8 + 0..7
__device__ __forceinline__ v16bf load_a_frag(const __bf16* rowk0, int hf) {
    v8bf lo = *(const v8bf*)(rowk0 + 16 * hf);
    v8bf hi = *(const v8bf*)(rowk0 + 16 * hf + 8);
    return __builtin_shufflevector(lo, hi, 0,1,2,3,4,5,6,7,8,9,10,11,12,13,14,15);
}

// ---------------- f32 -> bf16 conversion (vector x4) ----------------
__global__ void cvt_bf16_kernel(const float* __restrict__ in,
                                __bf16* __restrict__ out, int n4) {
    int i = blockIdx.x * blockDim.x + threadIdx.x;
    if (i >= n4) return;
    v4f v = *(const v4f*)(in + (size_t)i * 4);
    __bf16* o = out + (size_t)i * 4;
    o[0] = f2bf(v.x); o[1] = f2bf(v.y); o[2] = f2bf(v.z); o[3] = f2bf(v.w);
}

// ---------------- GEMM: C[M,N] = A[M,K] * B[N,K]^T, bf16 in / f32 out ----
// grid = (N/64, M/128), block = 256 (8 waves). Block tile 128x64.
// B tile (64 rows x 32 K, 4 KB) staged in LDS via gfx1250 async copy,
// double-buffered; A fragments register-pipelined. K-loop unrolled x2 with
// alternating registers/buffers (no register rotation, no hazard NOPs).
// Requires K % 64 == 0.
__global__ __launch_bounds__(256) void gemm_bf16_nt(const __bf16* __restrict__ A,
                                                    const __bf16* __restrict__ B,
                                                    float* __restrict__ C,
                                                    int M, int N, int K) {
    __shared__ __attribute__((aligned(64))) __bf16 sB[2][64 * 32];   // 2 x 4 KB

    const int tid  = threadIdx.x;
    const int lane = tid & 31;
    const int wave = tid >> 5;
    const int m0 = blockIdx.y * 128 + wave * 16;
    const int n0 = blockIdx.x * 64;
    const int mr = lane & 15;
    const int hf = lane >> 4;

    const __bf16* arow = A + (size_t)(m0 + mr) * K;

    // ---- async staging setup: thread tid copies 16 bytes of the 4 KB tile ----
    // LDS aperture flat address: low 32 bits == LDS byte offset (ISA 10.2).
    const unsigned ldsdst0 = (unsigned)(uintptr_t)(&sB[0][0]) + (unsigned)tid * 16u;
    const unsigned ldsdst1 = ldsdst0 + 64u * 32u * 2u;
    const int srow = tid >> 2, schunk = tid & 3;
    const unsigned gbase =
        (unsigned)(n0 + srow) * (unsigned)K * 2u + (unsigned)schunk * 16u;

    // prologue: stage k0 = 0 into buffer 0
    asm volatile("global_load_async_to_lds_b128 %0, %1, %2"
                 :: "v"(ldsdst0), "v"(gbase), "s"(B) : "memory");

    v16bf a0 = load_a_frag(arow, hf);
    v16bf a1;
    v8f acc0 = {}, acc1 = {}, acc2 = {}, acc3 = {};

    const __bf16* bb0 = &sB[0][0];
    const __bf16* bb1 = &sB[1][0];

    for (int k0 = 0; k0 < K; k0 += 64) {
        // ======== sub-step 0: consume buf0 (k0), stage k0+32 -> buf1 ========
        asm volatile("s_wait_asynccnt 0x0" ::: "memory");
        __syncthreads();
        {
            unsigned src = gbase + (unsigned)(k0 + 32) * 2u;   // K%64==0 -> always valid
            asm volatile("global_load_async_to_lds_b128 %0, %1, %2"
                         :: "v"(ldsdst1), "v"(src), "s"(B) : "memory");
        }
        __builtin_prefetch(arow + k0 + 256, 0, 3);
        a1 = load_a_frag(arow + k0 + 32, hf);

        {
            v16bf f0 = *(const v16bf*)(bb0 + ( 0 + mr) * 32 + 16 * hf);
            v16bf f1 = *(const v16bf*)(bb0 + (16 + mr) * 32 + 16 * hf);
            v16bf f2 = *(const v16bf*)(bb0 + (32 + mr) * 32 + 16 * hf);
            v16bf f3 = *(const v16bf*)(bb0 + (48 + mr) * 32 + 16 * hf);
            acc0 = wmma_bf16(a0, f0, acc0);
            acc1 = wmma_bf16(a0, f1, acc1);
            acc2 = wmma_bf16(a0, f2, acc2);
            acc3 = wmma_bf16(a0, f3, acc3);
        }

        // ======== sub-step 1: consume buf1 (k0+32), stage k0+64 -> buf0 =====
        asm volatile("s_wait_asynccnt 0x0" ::: "memory");
        __syncthreads();
        if (k0 + 64 < K) {
            unsigned src = gbase + (unsigned)(k0 + 64) * 2u;
            asm volatile("global_load_async_to_lds_b128 %0, %1, %2"
                         :: "v"(ldsdst0), "v"(src), "s"(B) : "memory");
            a0 = load_a_frag(arow + k0 + 64, hf);
        }

        {
            v16bf f0 = *(const v16bf*)(bb1 + ( 0 + mr) * 32 + 16 * hf);
            v16bf f1 = *(const v16bf*)(bb1 + (16 + mr) * 32 + 16 * hf);
            v16bf f2 = *(const v16bf*)(bb1 + (32 + mr) * 32 + 16 * hf);
            v16bf f3 = *(const v16bf*)(bb1 + (48 + mr) * 32 + 16 * hf);
            acc0 = wmma_bf16(a1, f0, acc0);
            acc1 = wmma_bf16(a1, f1, acc1);
            acc2 = wmma_bf16(a1, f2, acc2);
            acc3 = wmma_bf16(a1, f3, acc3);
        }
    }

    // D layout: VGPR r -> row m0 + r + 8*hf ; lane -> col
    #pragma unroll
    for (int r = 0; r < 8; ++r) {
        size_t row = (size_t)(m0 + r + 8 * hf) * N;
        C[row + n0 +  0 + mr] = acc0[r];
        C[row + n0 + 16 + mr] = acc1[r];
        C[row + n0 + 32 + mr] = acc2[r];
        C[row + n0 + 48 + mr] = acc3[r];
    }
}

// ---------------- RoPE-3D + reorganize q/k/v ----------------
// unit 0..15 : q head h  -> qout[(bt*16+h)*256 + s][d]        (rope)
// unit 16..19: k group g -> kout[(bt*4+g)*256 + s][d]         (rope)
// unit 20..23: v group g -> vT[(bt*4+g)*96 + d][s]            (transpose)
__global__ void rope_reorg_kernel(const float* __restrict__ qkv,
                                  __bf16* __restrict__ qout,
                                  __bf16* __restrict__ kout,
                                  __bf16* __restrict__ vT) {
    int idx = blockIdx.x * blockDim.x + threadIdx.x;
    if (idx >= MROWS * 24) return;
    int unit = idx % 24;
    int row  = idx / 24;
    int bt = row >> 8;
    int s  = row & 255;

    if (unit >= 20) {                      // v: transpose only
        int g = unit - 20;
        const float* src = qkv + (size_t)row * QKVOUT + NHEADS * HDIM + NGROUPS * HDIM + g * HDIM;
        __bf16* dst = vT + ((size_t)(bt * NGROUPS + g) * HDIM) * SS + s;
        #pragma unroll 8
        for (int d = 0; d < HDIM; ++d) dst[(size_t)d * SS] = f2bf(src[d]);
        return;
    }

    const float* src;
    __bf16* dst;
    if (unit < 16) {                       // q head
        src = qkv + (size_t)row * QKVOUT + unit * HDIM;
        dst = qout + ((size_t)(bt * NHEADS + unit) * SS + s) * HDIM;
    } else {                               // k group
        int g = unit - 16;
        src = qkv + (size_t)row * QKVOUT + NHEADS * HDIM + g * HDIM;
        dst = kout + ((size_t)(bt * NGROUPS + g) * SS + s) * HDIM;
    }

    float pos[3];
    pos[0] = (float)(bt % TT);             // t position
    pos[1] = (float)(s >> 4);              // h position
    pos[2] = (float)(s & 15);              // w position

    float x[HDIM];
    #pragma unroll 8
    for (int d = 0; d < HDIM; ++d) x[d] = src[d];

    #pragma unroll
    for (int seg = 0; seg < 3; ++seg) {
        int base = seg * 32;
        float p = pos[seg];
        #pragma unroll
        for (int j = 0; j < 16; ++j) {
            float invf = __powf(10000.0f, -(float)j * (1.0f / 16.0f));
            float ang = p * invf;
            float c = __cosf(ang), sn = __sinf(ang);
            float x1 = x[base + j], x2 = x[base + 16 + j];
            dst[base + j]      = f2bf(x1 * c - x2 * sn);
            dst[base + 16 + j] = f2bf(x1 * sn + x2 * c);
        }
    }
}

// ---------------- attention: one wave per (bh, 16-row q tile) ----------
// grid = (512, 16), block = 32
__global__ __launch_bounds__(32) void attn_kernel(const __bf16* __restrict__ q,
                                                  const __bf16* __restrict__ k,
                                                  const __bf16* __restrict__ vT,
                                                  const int* __restrict__ pmask,
                                                  __bf16* __restrict__ attn_out) {
    __shared__ float sS[16 * 256];
    __shared__ float redm[2][16];
    __shared__ float redsum[2][16];

    const int lane = threadIdx.x;
    const int bh = blockIdx.x;             // 0..511
    const int q0 = blockIdx.y * 16;
    const int bt = bh >> 4;
    const int h  = bh & 15;
    const int bg = bt * NGROUPS + (h >> 2);
    const int mr = lane & 15;
    const int hf = lane >> 4;
    const float scale = 0.1020620726159658f;   // 1/sqrt(96)

    const __bf16* qbase = q + ((size_t)bh * SS + q0 + mr) * HDIM;

    // ---- scores = q . k^T  (16 x 256) into LDS ----
    for (int kt = 0; kt < SS; kt += 16) {
        const __bf16* kbase = k + ((size_t)bg * SS + kt + mr) * HDIM + 16 * hf;
        v8f acc = {};
        #pragma unroll
        for (int d0 = 0; d0 < HDIM; d0 += 32) {
            v16bf a = load_a_frag(qbase + d0, hf);
            v16bf b = *(const v16bf*)(kbase + d0);
            acc = wmma_bf16(a, b, acc);
        }
        int pm = pmask[bt * SS + kt + mr];
        #pragma unroll
        for (int r = 0; r < 8; ++r) {
            float sv = pm ? acc[r] * scale : -1.0e9f;
            sS[(r + 8 * hf) * 256 + kt + mr] = sv;
        }
    }
    __syncthreads();

    // ---- softmax over 256 keys per row; lane -> (row=mr, half=hf) ----
    {
        float* rowp = sS + mr * 256 + hf * 128;
        float mx = -3.4e38f;
        for (int i = 0; i < 128; ++i) mx = fmaxf(mx, rowp[i]);
        redm[hf][mr] = mx;
        __syncthreads();
        float mfull = fmaxf(redm[0][mr], redm[1][mr]);
        float ssum = 0.0f;
        for (int i = 0; i < 128; ++i) {
            float e = __expf(rowp[i] - mfull);
            rowp[i] = e;
            ssum += e;
        }
        redsum[hf][mr] = ssum;
        __syncthreads();
        float inv = 1.0f / (redsum[0][mr] + redsum[1][mr]);
        for (int i = 0; i < 128; ++i) rowp[i] *= inv;
    }
    __syncthreads();

    // ---- out = P . V   (16 x 96), V pre-transposed [d][kpos] ----
    v8f oacc[6];
    #pragma unroll
    for (int j = 0; j < 6; ++j) oacc[j] = (v8f){};

    for (int kk = 0; kk < SS; kk += 32) {
        v16bf a;
        #pragma unroll
        for (int i = 0; i < 16; ++i) {
            int kidx = kk + 16 * hf + i;
            a[i] = f2bf(sS[mr * 256 + kidx]);
        }
        #pragma unroll
        for (int j = 0; j < 6; ++j) {
            const __bf16* vb = vT + ((size_t)bg * HDIM + j * 16 + mr) * SS + kk + 16 * hf;
            v16bf b = *(const v16bf*)vb;
            oacc[j] = wmma_bf16(a, b, oacc[j]);
        }
    }
    #pragma unroll
    for (int j = 0; j < 6; ++j)
        #pragma unroll
        for (int r = 0; r < 8; ++r)
            attn_out[((size_t)bt * SS + q0 + r + 8 * hf) * DMODEL + h * HDIM + j * 16 + mr]
                = f2bf(oacc[j][r]);
}

// ---------------- launch ----------------
extern "C" void kernel_launch(void* const* d_in, const int* in_sizes, int n_in,
                              void* d_out, int out_size, void* d_ws, size_t ws_size,
                              hipStream_t stream) {
    const float* x      = (const float*)d_in[0];
    const int*   pmask  = (const int*)d_in[1];
    const float* w_qkv  = (const float*)d_in[2];
    const float* w_o    = (const float*)d_in[3];
    float*       out    = (float*)d_out;

    char* ws = (char*)d_ws;
    size_t off = 0;
    auto carve = [&](size_t bytes) -> char* {
        char* p = ws + off;
        off = (off + bytes + 255) & ~(size_t)255;
        return p;
    };

    const size_t nX   = (size_t)MROWS * DMODEL;        // 12,582,912
    const size_t nWq  = (size_t)QKVOUT * DMODEL;
    const size_t nWo  = (size_t)DMODEL * DMODEL;

    __bf16* x_bf    = (__bf16*)carve(nX * 2);
    __bf16* wq_bf   = (__bf16*)carve(nWq * 2);
    __bf16* wo_bf   = (__bf16*)carve(nWo * 2);
    float*  qkv_f   = (float*) carve((size_t)MROWS * QKVOUT * 4);
    __bf16* q_bf    = (__bf16*)carve((size_t)BB * TT * NHEADS  * SS * HDIM * 2);
    __bf16* k_bf    = (__bf16*)carve((size_t)BB * TT * NGROUPS * SS * HDIM * 2);
    __bf16* vT_bf   = (__bf16*)carve((size_t)BB * TT * NGROUPS * HDIM * SS * 2);
    __bf16* attn_bf = (__bf16*)carve(nX * 2);

    // 1) convert operands to bf16
    cvt_bf16_kernel<<<dim3((unsigned)(nX  / 4 / 256)), 256, 0, stream>>>(x,     x_bf,  (int)(nX  / 4));
    cvt_bf16_kernel<<<dim3((unsigned)(nWq / 4 / 256)), 256, 0, stream>>>(w_qkv, wq_bf, (int)(nWq / 4));
    cvt_bf16_kernel<<<dim3((unsigned)(nWo / 4 / 256)), 256, 0, stream>>>(w_o,   wo_bf, (int)(nWo / 4));

    // 2) QKV projection: qkv[8192,2304] = x[8192,1536] * w_qkv[2304,1536]^T
    gemm_bf16_nt<<<dim3(QKVOUT / 64, MROWS / 128), 256, 0, stream>>>(
        x_bf, wq_bf, qkv_f, MROWS, QKVOUT, DMODEL);

    // 3) RoPE-3D + layout for attention
    rope_reorg_kernel<<<dim3(MROWS * 24 / 256), 256, 0, stream>>>(qkv_f, q_bf, k_bf, vT_bf);

    // 4) attention (512 bh x 16 q-tiles)
    attn_kernel<<<dim3(BB * TT * NHEADS, SS / 16), 32, 0, stream>>>(
        q_bf, k_bf, vT_bf, pmask, attn_bf);

    // 5) output projection: out[8192,1536] = attn[8192,1536] * w_o[1536,1536]^T
    gemm_bf16_nt<<<dim3(DMODEL / 64, MROWS / 128), 256, 0, stream>>>(
        attn_bf, wo_bf, out, MROWS, DMODEL, DMODEL);
}
